// Decoder_64982855188927
// MI455X (gfx1250) — matrix operand order repaired and checked by reading, not compiled, not run on hardware
//
#include <hip/hip_runtime.h>
#include <hip/hip_bf16.h>

typedef __attribute__((ext_vector_type(16))) _Float16 v16h;
typedef __attribute__((ext_vector_type(8)))  _Float16 v8h;
typedef __attribute__((ext_vector_type(8)))  float    v8f;

#define N_CLASSES 10
#define UNIT      16
#define DIN       160     // 10*16
#define H1N       512
#define H2N       1024
#define OUTN      3072
#define BATCH     32768

// ---------------------------------------------------------------------------
// Kernel 1: capsule mask.  One thread per sample: argmax of capsule L2 norms,
// write masked 160-vector as f16 (input A0 of the first GEMM).
// ---------------------------------------------------------------------------
__global__ __launch_bounds__(256) void mask_kernel(const float* __restrict__ x,
                                                   _Float16* __restrict__ a0) {
  const int tid = blockIdx.x * 256 + threadIdx.x;
  const float4* row = reinterpret_cast<const float4*>(x + (size_t)tid * DIN);

  float best = -1.0f;
  int bj = 0;
#pragma unroll
  for (int j = 0; j < N_CLASSES; ++j) {
    float s = 0.0f;
#pragma unroll
    for (int q = 0; q < 4; ++q) {
      float4 v = row[j * 4 + q];
      s += v.x * v.x + v.y * v.y + v.z * v.z + v.w * v.w;
    }
    if (s > best) { best = s; bj = j; }   // first max wins (matches jnp.argmax)
  }

  const float* win = x + (size_t)tid * DIN + bj * UNIT;
  v8h* out = reinterpret_cast<v8h*>(a0 + (size_t)tid * DIN);
#pragma unroll
  for (int j = 0; j < N_CLASSES; ++j) {
#pragma unroll
    for (int h = 0; h < 2; ++h) {
      v8h o;
#pragma unroll
      for (int e = 0; e < 8; ++e)
        o[e] = (j == bj) ? (_Float16)win[h * 8 + e] : (_Float16)0.0f;
      out[j * 2 + h] = o;
    }
  }
}

// ---------------------------------------------------------------------------
// Kernel 2: convert fp32 weights [K,N] -> f16 transposed [N,K].
// Transposed layout makes the WMMA B-fragment a contiguous per-lane load.
// ---------------------------------------------------------------------------
__global__ __launch_bounds__(256) void wconv_kernel(const float* __restrict__ W,
                                                    _Float16* __restrict__ Wt,
                                                    int K, int N) {
  const int idx = blockIdx.x * 256 + threadIdx.x;
  if (idx >= K * N) return;
  const int n = idx / K;
  const int k = idx - n * K;
  Wt[idx] = (_Float16)W[(size_t)k * N + n];
}

// ---------------------------------------------------------------------------
// Kernel 3: WMMA GEMM  out = act(A[M,K] @ W[K,N] + b),  W given as Wt[N,K] f16.
// Wave32.  Each wave computes a 64x64 tile: 4 M-subtiles x 4 N-subtiles
// (16 v8f accumulators).  Per K-step (32): 4 A-frags + 4 B-frags -> 16 WMMAs,
// i.e. 4x reuse of both operands and 16 independent WMMA chains for latency
// hiding.  Block = 8 waves as 2(M) x 4(N) -> 128 x 256 per block.
// ACT: 0 = ReLU -> f16 store (hidden layers), 1 = sigmoid -> f32 store (final).
// ---------------------------------------------------------------------------
template <int ACT>
__global__ __launch_bounds__(256) void gemm_kernel(const _Float16* __restrict__ A,
                                                   const _Float16* __restrict__ Wt,
                                                   const float* __restrict__ bias,
                                                   void* __restrict__ outp,
                                                   int K, int N) {
  const int lane = threadIdx.x & 31;
  const int wave = threadIdx.x >> 5;
  const int wm   = wave >> 2;                         // 0..1  (M direction)
  const int wn   = wave & 3;                          // 0..3  (N direction)
  const int hi   = lane >> 4;
  const int mr   = lane & 15;

  const int m0 = blockIdx.x * 128 + wm * 64;          // wave's M base (4 tiles)
  const int n0 = blockIdx.y * 256 + wn * 64;          // wave's N base (4 tiles)

  v8f acc[4][4] = {};                                 // [tm][tn] 16x16 f32 tiles

  // A fragment (ISA 16-bit A 16x32 layout): lane holds row mr of its subtile,
  // e=0..7 -> K = hi*8 + e ; e=8..15 -> K = 16 + hi*8 + (e-8).
  const _Float16* arow[4];
#pragma unroll
  for (int tm = 0; tm < 4; ++tm)
    arow[tm] = A + (size_t)(m0 + tm * 16 + mr) * K + hi * 8;

  // B fragment (ISA 16-bit B 32x16 layout): lane holds col mr of its subtile,
  // e=0..7 -> K = hi*8 + e ; e=8..15 -> K = 16 + hi*8 + (e-8).
  const _Float16* brow[4];
#pragma unroll
  for (int tn = 0; tn < 4; ++tn)
    brow[tn] = Wt + (size_t)(n0 + tn * 16 + mr) * K + hi * 8;

  for (int k0 = 0; k0 < K; k0 += 32) {
    // Load all 8 fragments for this K-step first (clause-friendly), then
    // run 16 independent WMMAs.
    v16h afrag[4];
#pragma unroll
    for (int tm = 0; tm < 4; ++tm) {
      v8h alo = *reinterpret_cast<const v8h*>(arow[tm] + k0);
      v8h ahi = *reinterpret_cast<const v8h*>(arow[tm] + k0 + 16);
      afrag[tm] = __builtin_shufflevector(alo, ahi,
          0, 1, 2, 3, 4, 5, 6, 7, 8, 9, 10, 11, 12, 13, 14, 15);
    }
    v16h bfrag[4];
#pragma unroll
    for (int tn = 0; tn < 4; ++tn) {
      v8h blo = *reinterpret_cast<const v8h*>(brow[tn] + k0);
      v8h bhi = *reinterpret_cast<const v8h*>(brow[tn] + k0 + 16);
      bfrag[tn] = __builtin_shufflevector(blo, bhi,
          0, 1, 2, 3, 4, 5, 6, 7, 8, 9, 10, 11, 12, 13, 14, 15);
    }
#pragma unroll
    for (int tn = 0; tn < 4; ++tn) {
#pragma unroll
      for (int tm = 0; tm < 4; ++tm) {
        acc[tm][tn] = __builtin_amdgcn_wmma_f32_16x16x32_f16(
            /*neg_a=*/false, afrag[tm], /*neg_b=*/false, bfrag[tn],
            /*c_mod=*/(short)0, acc[tm][tn],
            /*reuse_a=*/false, /*reuse_b=*/false);
      }
    }
  }

  // Epilogue: C/D layout = VGPR r -> row r + 8*hi, lane mr -> column.
#pragma unroll
  for (int tn = 0; tn < 4; ++tn) {
    const int col = n0 + tn * 16 + mr;
    const float bv = bias[col];
#pragma unroll
    for (int tm = 0; tm < 4; ++tm) {
#pragma unroll
      for (int r = 0; r < 8; ++r) {
        const int row = m0 + tm * 16 + r + 8 * hi;
        float v = acc[tm][tn][r] + bv;
        if (ACT == 0) {
          v = fmaxf(v, 0.0f);
          reinterpret_cast<_Float16*>(outp)[(size_t)row * N + col] = (_Float16)v;
        } else {
          v = 1.0f / (1.0f + __expf(-v));
          reinterpret_cast<float*>(outp)[(size_t)row * N + col] = v;
        }
      }
    }
  }
}

// ---------------------------------------------------------------------------
extern "C" void kernel_launch(void* const* d_in, const int* in_sizes, int n_in,
                              void* d_out, int out_size, void* d_ws, size_t ws_size,
                              hipStream_t stream) {
  const float* x  = (const float*)d_in[0];
  // d_in[1] = target (unused by forward)
  const float* W1 = (const float*)d_in[2];
  const float* b1 = (const float*)d_in[3];
  const float* W2 = (const float*)d_in[4];
  const float* b2 = (const float*)d_in[5];
  const float* W3 = (const float*)d_in[6];
  const float* b3 = (const float*)d_in[7];
  float* out = (float*)d_out;

  // Workspace carve-out (f16 buffers, 256B aligned).
  char* ws = (char*)d_ws;
  size_t off = 0;
  auto carve = [&](size_t elems) -> _Float16* {
    _Float16* p = (_Float16*)(ws + off);
    off += ((elems * sizeof(_Float16) + 255) / 256) * 256;
    return p;
  };
  _Float16* A0  = carve((size_t)BATCH * DIN);   // masked input, f16
  _Float16* H1v = carve((size_t)BATCH * H1N);   // hidden 1, f16
  _Float16* H2v = carve((size_t)BATCH * H2N);   // hidden 2, f16
  _Float16* W1t = carve((size_t)DIN * H1N);     // W1^T f16 [512,160]
  _Float16* W2t = carve((size_t)H1N * H2N);     // W2^T f16 [1024,512]
  _Float16* W3t = carve((size_t)H2N * OUTN);    // W3^T f16 [3072,1024]

  // 1) capsule mask -> f16 A0
  mask_kernel<<<BATCH / 256, 256, 0, stream>>>(x, A0);

  // 2) weight convert + transpose (tiny vs GEMM cost; weights stay L2-hot)
  wconv_kernel<<<(DIN * H1N + 255) / 256, 256, 0, stream>>>(W1, W1t, DIN, H1N);
  wconv_kernel<<<(H1N * H2N + 255) / 256, 256, 0, stream>>>(W2, W2t, H1N, H2N);
  wconv_kernel<<<(H2N * OUTN + 255) / 256, 256, 0, stream>>>(W3, W3t, H2N, OUTN);

  // 3) three WMMA GEMM layers (block tile 128 x 256)
  dim3 g1(BATCH / 128, H1N / 256);
  gemm_kernel<0><<<g1, 256, 0, stream>>>(A0, W1t, b1, H1v, DIN, H1N);
  dim3 g2(BATCH / 128, H2N / 256);
  gemm_kernel<0><<<g2, 256, 0, stream>>>(H1v, W2t, b2, H2v, H1N, H2N);
  dim3 g3(BATCH / 128, OUTN / 256);
  gemm_kernel<1><<<g3, 256, 0, stream>>>(H2v, W3t, b3, out, H2N, OUTN);
}
